// MultiHeadAttention_65309272703565
// MI455X (gfx1250) — matrix operand order repaired and checked by reading, not compiled, not run on hardware
//
#include <hip/hip_runtime.h>
#include <hip/hip_bf16.h>

// Problem constants (from reference)
#define BB   4
#define SS   2048
#define EE   512
#define HH   8
#define DHD  64
#define N3   1536   // 3*E
#define MTOT 8192   // B*S

typedef __attribute__((ext_vector_type(16))) __bf16 v16bf;
typedef __attribute__((ext_vector_type(8)))  __bf16 v8bf;
typedef __attribute__((ext_vector_type(8)))  float  v8f;

static __device__ __forceinline__ v16bf cat8(v8bf lo, v8bf hi) {
    return __builtin_shufflevector(lo, hi, 0,1,2,3,4,5,6,7,8,9,10,11,12,13,14,15);
}

static __device__ __forceinline__ unsigned pack_bf16x2(float a, float b) {
    __bf16 x = (__bf16)a, y = (__bf16)b;
    unsigned short xu = __builtin_bit_cast(unsigned short, x);
    unsigned short yu = __builtin_bit_cast(unsigned short, y);
    return (unsigned)xu | ((unsigned)yu << 16);
}

static __device__ __forceinline__ float fexp2(float x) {
#if __has_builtin(__builtin_amdgcn_exp2f)
    return __builtin_amdgcn_exp2f(x);
#else
    return exp2f(x);
#endif
}

static __device__ __forceinline__ v8f wmma_bf16(v16bf a, v16bf b, v8f c) {
    return __builtin_amdgcn_wmma_f32_16x16x32_bf16(
        /*neg_a=*/false, a, /*neg_b=*/false, b,
        /*c_mod=*/(short)0, c, /*reuse_a=*/false, /*reuse_b=*/false);
}

// Low 32 bits of a generic shared-memory pointer == LDS byte offset
// (LDS aperture mapping).
static __device__ __forceinline__ unsigned lds_addr32(const void* p) {
    return (unsigned)(unsigned long long)p;
}

// GVS-mode async copy: LDS[vdst] <- MEM[saddr + vaddr], 16 bytes per lane.
// Tracked by ASYNCcnt.
static __device__ __forceinline__ void async_ld_b128(unsigned ldsa,
                                                     unsigned long long gbase,
                                                     unsigned goff) {
    asm volatile("global_load_async_to_lds_b128 %0, %1, %2"
                 :: "v"(ldsa), "v"(goff), "s"(gbase)
                 : "memory");
}

static __device__ __forceinline__ void wait_async0() {
    asm volatile("s_wait_asynccnt 0x0" ::: "memory");
}

// ---------------------------------------------------------------------------
// Kernel 1: QKV projection  out[m,f] = sum_k X[m,k]*W[f,k] + bias[f]
// Writes bf16 Q,K as [B,H,S,64] and V transposed as [B,H,64,S] to workspace.
// Block: 256 thr (8 waves), tile 128(M) x 128(N), K staged 64-wide in LDS.
// ---------------------------------------------------------------------------
__global__ __launch_bounds__(256) void qkv_proj_kernel(
    const float* __restrict__ X, const float* __restrict__ W,
    const float* __restrict__ bias,
    __bf16* __restrict__ Qo, __bf16* __restrict__ Ko, __bf16* __restrict__ Vt)
{
    __shared__ __align__(16) __bf16 At[128 * 72];   // [row][k], pad 64->72
    __shared__ __align__(16) __bf16 Bt[128 * 72];   // [n][k],   pad 64->72

    const int tid  = threadIdx.x;
    const int lane = tid & 31;
    const int wave = tid >> 5;
    const int wm   = wave >> 1;          // 0..3  (32 rows each)
    const int wn   = wave & 1;           // 0..1  (64 cols each)
    const int m0   = blockIdx.y * 128;
    const int n0   = blockIdx.x * 128;

    v8f acc[2][4];
#pragma unroll
    for (int i = 0; i < 2; ++i)
#pragma unroll
        for (int j = 0; j < 4; ++j)
#pragma unroll
            for (int r = 0; r < 8; ++r) acc[i][j][r] = 0.f;

    for (int kc = 0; kc < EE; kc += 64) {
        // Stage A (128x64 f32 -> bf16) and B (128x64 f32 -> bf16)
#pragma unroll
        for (int p = 0; p < 8; ++p) {
            int idx = tid + p * 256;
            int row = idx >> 4;
            int c4  = idx & 15;
            float4 fa = *(const float4*)(X + (size_t)(m0 + row) * EE + kc + c4 * 4);
            unsigned* da = (unsigned*)(At + row * 72 + c4 * 4);
            da[0] = pack_bf16x2(fa.x, fa.y);
            da[1] = pack_bf16x2(fa.z, fa.w);
            float4 fb = *(const float4*)(W + (size_t)(n0 + row) * EE + kc + c4 * 4);
            unsigned* db = (unsigned*)(Bt + row * 72 + c4 * 4);
            db[0] = pack_bf16x2(fb.x, fb.y);
            db[1] = pack_bf16x2(fb.z, fb.w);
        }
        __syncthreads();

#pragma unroll
        for (int kk = 0; kk < 64; kk += 32) {
            v16bf af[2], bf[4];
#pragma unroll
            for (int ti = 0; ti < 2; ++ti) {
                int r  = wm * 32 + ti * 16 + (lane & 15);
                int kb = kk + ((lane >> 4) << 3);            // A layout
                v8bf lo = *(const v8bf*)(At + r * 72 + kb);
                v8bf hi = *(const v8bf*)(At + r * 72 + kb + 16);
                af[ti] = cat8(lo, hi);
            }
#pragma unroll
            for (int tj = 0; tj < 4; ++tj) {
                int n  = wn * 64 + tj * 16 + (lane & 15);
                int kb = kk + ((lane >> 4) << 3);            // B layout
                v8bf lo = *(const v8bf*)(Bt + n * 72 + kb);
                v8bf hi = *(const v8bf*)(Bt + n * 72 + kb + 16);
                bf[tj] = cat8(lo, hi);
            }
#pragma unroll
            for (int ti = 0; ti < 2; ++ti)
#pragma unroll
                for (int tj = 0; tj < 4; ++tj)
                    acc[ti][tj] = wmma_bf16(af[ti], bf[tj], acc[ti][tj]);
        }
        __syncthreads();
    }

    // Epilogue: bias + scatter to Q / K / V^T workspace (bf16)
#pragma unroll
    for (int tj = 0; tj < 4; ++tj) {
        int n   = n0 + wn * 64 + tj * 16 + (lane & 15);
        float bv = bias[n];
        int sel = n >> 9;        // 0=Q 1=K 2=V
        int e   = n & 511;
        int h   = e >> 6;
        int d   = e & 63;
#pragma unroll
        for (int ti = 0; ti < 2; ++ti) {
#pragma unroll
            for (int r = 0; r < 8; ++r) {
                int m = m0 + wm * 32 + ti * 16 + r + ((lane >> 4) << 3);
                int b = m >> 11;           // / S
                int s = m & 2047;          // % S
                __bf16 o = (__bf16)(acc[ti][tj][r] + bv);
                if (sel == 0)
                    Qo[(((size_t)b * HH + h) * SS + s) * DHD + d] = o;
                else if (sel == 1)
                    Ko[(((size_t)b * HH + h) * SS + s) * DHD + d] = o;
                else
                    Vt[(((size_t)b * HH + h) * DHD + d) * SS + s] = o;
            }
        }
    }
}

// ---------------------------------------------------------------------------
// Kernel 2: flash attention. Grid (S/128, H, B), 256 threads (8 waves).
// Wave w owns Q rows [16w,16w+16). 128-key tiles, online softmax.
// K/V tiles are double-buffered and streamed into LDS with
// global_load_async_to_lds_b128 (ASYNCcnt), overlapping with WMMA compute.
// scale = 1/sqrt(E) per the reference.
// ---------------------------------------------------------------------------
__global__ __launch_bounds__(256) void attn_kernel(
    const __bf16* __restrict__ Qi, const __bf16* __restrict__ Ki,
    const __bf16* __restrict__ Vt, float* __restrict__ Out)
{
    __shared__ __align__(16) __bf16 sQ[128 * 72];       // [row][d]  pad 64->72
    __shared__ __align__(16) __bf16 sK[2][128 * 72];    // [key][d]  pad 64->72
    __shared__ __align__(16) __bf16 sV[2][64 * 136];    // [d][key]  pad 128->136
    __shared__ __align__(16) __bf16 sP[128 * 136];      // [row][key] pad 128->136

    const int tid  = threadIdx.x;
    const int lane = tid & 31;
    const int wave = tid >> 5;
    const int sq0  = blockIdx.x * 128;
    const int h    = blockIdx.y;
    const int b    = blockIdx.z;
    const size_t bh = (size_t)b * HH + h;

    // Issue async K/V tile loads for key-tile jt into buffer `buf`.
    auto issue_tile = [&](int jt, int buf) {
        const unsigned long long kb =
            (unsigned long long)(const void*)(Ki + (bh * SS + (size_t)jt * 128) * DHD);
        const unsigned long long vb =
            (unsigned long long)(const void*)(Vt + bh * DHD * SS + (size_t)jt * 128);
#pragma unroll
        for (int p = 0; p < 4; ++p) {
            int idx = tid + p * 256;
            int row = idx >> 3;          // key in [0,128)
            int ch  = idx & 7;           // 8 x 16B per key row
            async_ld_b128(lds_addr32(&sK[buf][row * 72 + ch * 8]),
                          kb, (unsigned)(row * 128 + ch * 16));
        }
#pragma unroll
        for (int p = 0; p < 4; ++p) {
            int idx = tid + p * 256;
            int row = idx >> 4;          // d in [0,64)
            int ch  = idx & 15;          // 16 x 16B per d row
            async_ld_b128(lds_addr32(&sV[buf][row * 136 + ch * 8]),
                          vb, (unsigned)(row * (SS * 2) + ch * 16));
        }
    };

    // Load Q tile (bf16, already converted) + kick off first K/V tile
    const __bf16* qbase = Qi + (bh * SS + sq0) * DHD;
#pragma unroll
    for (int p = 0; p < 4; ++p) {
        int idx = tid + p * 256;
        int row = idx >> 3;
        int ch  = idx & 7;
        *(uint4*)(sQ + row * 72 + ch * 8) = *(const uint4*)(qbase + (size_t)row * DHD + ch * 8);
    }
    issue_tile(0, 0);
    __syncthreads();

    // Persistent Q A-fragments (dh=64 -> 2 k-steps)
    v16bf aq[2];
    {
        int r = wave * 16 + (lane & 15);
#pragma unroll
        for (int kk2 = 0; kk2 < 2; ++kk2) {
            int kb = kk2 * 32 + ((lane >> 4) << 3);
            v8bf lo = *(const v8bf*)(sQ + r * 72 + kb);
            v8bf hi = *(const v8bf*)(sQ + r * 72 + kb + 16);
            aq[kk2] = cat8(lo, hi);
        }
    }

    v8f o[4];
    float mrow[8], lrow[8];
#pragma unroll
    for (int t = 0; t < 4; ++t)
#pragma unroll
        for (int r = 0; r < 8; ++r) o[t][r] = 0.f;
#pragma unroll
    for (int r = 0; r < 8; ++r) { mrow[r] = -1e30f; lrow[r] = 0.f; }

    const float k1 = 0.044194173824159216f * 1.4426950408889634f; // (1/sqrt(512))*log2(e)

    for (int j = 0; j < SS / 128; ++j) {
        const int cur = j & 1;
        const __bf16* __restrict__ kt = sK[cur];
        const __bf16* __restrict__ vt = sV[cur];

        wait_async0();       // this wave's tile-j async copies have landed
        __syncthreads();     // everyone's copies landed; prev buffer fully consumed
        if (j + 1 < SS / 128) issue_tile(j + 1, cur ^ 1);   // overlap next tile

        // S = Q K^T : 16 rows x 128 keys per wave
        v8f sacc[8];
#pragma unroll
        for (int tj = 0; tj < 8; ++tj)
#pragma unroll
            for (int r = 0; r < 8; ++r) sacc[tj][r] = 0.f;

#pragma unroll
        for (int tj = 0; tj < 8; ++tj) {
#pragma unroll
            for (int kk2 = 0; kk2 < 2; ++kk2) {
                int n  = tj * 16 + (lane & 15);
                int kb = kk2 * 32 + ((lane >> 4) << 3);
                v8bf lo = *(const v8bf*)(kt + n * 72 + kb);
                v8bf hi = *(const v8bf*)(kt + n * 72 + kb + 16);
                sacc[tj] = wmma_bf16(aq[kk2], cat8(lo, hi), sacc[tj]);
            }
        }

        // Online softmax (rows live in 16-lane halves of the wave)
#pragma unroll
        for (int r = 0; r < 8; ++r) {
            float mx = sacc[0][r];
#pragma unroll
            for (int tj = 1; tj < 8; ++tj) mx = fmaxf(mx, sacc[tj][r]);
#pragma unroll
            for (int off = 1; off < 16; off <<= 1)
                mx = fmaxf(mx, __shfl_xor(mx, off, 32));
            mx *= k1;
            float mn    = fmaxf(mrow[r], mx);
            float alpha = fexp2(mrow[r] - mn);
            mrow[r] = mn;
            float rs = 0.f;
#pragma unroll
            for (int tj = 0; tj < 8; ++tj) {
                float pv = fexp2(sacc[tj][r] * k1 - mn);
                sacc[tj][r] = pv;
                rs += pv;
            }
#pragma unroll
            for (int off = 1; off < 16; off <<= 1)
                rs += __shfl_xor(rs, off, 32);
            lrow[r] = lrow[r] * alpha + rs;
#pragma unroll
            for (int t = 0; t < 4; ++t) o[t][r] *= alpha;
        }

        // P -> LDS (bf16). Wave-private rows: LDS in-order per wave, no barrier.
        {
            int rowb = wave * 16 + ((lane >> 4) << 3);
#pragma unroll
            for (int r = 0; r < 8; ++r)
#pragma unroll
                for (int tj = 0; tj < 8; ++tj)
                    sP[(rowb + r) * 136 + tj * 16 + (lane & 15)] = (__bf16)sacc[tj][r];
        }

        // O += P @ V  (K-dim = 128 keys -> 4 k-steps)
#pragma unroll
        for (int kk4 = 0; kk4 < 4; ++kk4) {
            int kofs = kk4 * 32;
            v16bf ap;
            {
                int r  = wave * 16 + (lane & 15);
                int kb = kofs + ((lane >> 4) << 3);
                v8bf lo = *(const v8bf*)(sP + r * 136 + kb);
                v8bf hi = *(const v8bf*)(sP + r * 136 + kb + 16);
                ap = cat8(lo, hi);
            }
#pragma unroll
            for (int t = 0; t < 4; ++t) {
                int n  = t * 16 + (lane & 15);
                int kb = kofs + ((lane >> 4) << 3);
                v8bf lo = *(const v8bf*)(vt + n * 136 + kb);
                v8bf hi = *(const v8bf*)(vt + n * 136 + kb + 16);
                o[t] = wmma_bf16(ap, cat8(lo, hi), o[t]);
            }
        }
        __syncthreads();   // all waves done with sP/current buffers this iter
    }

    // Epilogue: out[b, s, h*64 + d] = O / l  (fp32)
#pragma unroll
    for (int r = 0; r < 8; ++r) {
        float inv = 1.0f / lrow[r];
        int row = wave * 16 + r + ((lane >> 4) << 3);
        int s   = sq0 + row;
        float* op = Out + ((size_t)b * SS + s) * EE + h * DHD;
#pragma unroll
        for (int t = 0; t < 4; ++t)
            op[t * 16 + (lane & 15)] = o[t][r] * inv;
    }
}

// ---------------------------------------------------------------------------
extern "C" void kernel_launch(void* const* d_in, const int* in_sizes, int n_in,
                              void* d_out, int out_size, void* d_ws, size_t ws_size,
                              hipStream_t stream) {
    (void)in_sizes; (void)n_in; (void)out_size; (void)ws_size;
    const float* X    = (const float*)d_in[0];   // [B,S,E]
    const float* W    = (const float*)d_in[1];   // [3E,E]
    const float* bias = (const float*)d_in[2];   // [3E]
    float* out = (float*)d_out;                  // [B,S,E]

    const size_t QS = (size_t)BB * HH * SS * DHD;   // 4,194,304 elements
    __bf16* q  = (__bf16*)d_ws;
    __bf16* k  = q + QS;
    __bf16* vt = k + QS;                            // 24 MB total bf16 scratch

    qkv_proj_kernel<<<dim3(N3 / 128, MTOT / 128), 256, 0, stream>>>(X, W, bias, q, k, vt);
    attn_kernel<<<dim3(SS / 128, HH, BB), 256, 0, stream>>>(q, k, vt, out);
}
